// EdgeNet_55542517072002
// MI455X (gfx1250) — compile-verified, matrix operand order checked by reading, not run on hardware
//
#include <hip/hip_runtime.h>
#include <hip/hip_bf16.h>

#define D       256
#define PAD     8            // bf16 elements of row padding -> 528B row stride,
                             // lane r hits banks 4r..4r+3: conflict-free ds_load_b128
#define STRIDE  (D + PAD)    // 264 elements
#define M_TILE  128
#define BLOCK   256          // 8 waves of 32

typedef __attribute__((ext_vector_type(16))) __bf16       v16bf;
typedef __attribute__((ext_vector_type(8)))  float        v8f;
typedef __attribute__((ext_vector_type(8)))  unsigned int v8u;

// fp32 -> bf16 round-to-nearest-even, two packed into a dword (low = first)
static __device__ __forceinline__ unsigned int pack_bf2(float a, float b) {
    unsigned int ua = __float_as_uint(a);
    unsigned int ub = __float_as_uint(b);
    ua = (ua + 0x7FFFu + ((ua >> 16) & 1u)) >> 16;
    ub = (ub + 0x7FFFu + ((ub >> 16) & 1u)) >> 16;
    return ua | (ub << 16);
}

static __device__ __forceinline__ unsigned short f2bf(float a) {
    unsigned int ua = __float_as_uint(a);
    return (unsigned short)((ua + 0x7FFFu + ((ua >> 16) & 1u)) >> 16);
}

// A-fragment (16x32 bf16) from padded row-major bf16 LDS buffer.
// elemBase = row*STRIDE + k0 + hi*8 ; loads K chunk [0..7] and [+16..+23]
static __device__ __forceinline__ v16bf load_frag_lds(const unsigned short* s, int elemBase) {
    uint4 x0 = *(const uint4*)(s + elemBase);
    uint4 x1 = *(const uint4*)(s + elemBase + 16);
    v8u u;
    u[0] = x0.x; u[1] = x0.y; u[2] = x0.z; u[3] = x0.w;
    u[4] = x1.x; u[5] = x1.y; u[6] = x1.z; u[7] = x1.w;
    return __builtin_bit_cast(v16bf, u);
}

// B-fragment (32x16 bf16) straight from a row-major fp32 weight row (out = x @ W^T,
// so B[k][n] = W[n][k]; lane's column row pointer is passed in). kbase = k0 + hi*8.
static __device__ __forceinline__ v16bf load_bfrag_global(const float* __restrict__ wrow, int kbase) {
    float4 f0 = ((const float4*)(wrow + kbase))[0];
    float4 f1 = ((const float4*)(wrow + kbase))[1];
    float4 f2 = ((const float4*)(wrow + kbase + 16))[0];
    float4 f3 = ((const float4*)(wrow + kbase + 16))[1];
    v8u u;
    u[0] = pack_bf2(f0.x, f0.y); u[1] = pack_bf2(f0.z, f0.w);
    u[2] = pack_bf2(f1.x, f1.y); u[3] = pack_bf2(f1.z, f1.w);
    u[4] = pack_bf2(f2.x, f2.y); u[5] = pack_bf2(f2.z, f2.w);
    u[6] = pack_bf2(f3.x, f3.y); u[7] = pack_bf2(f3.z, f3.w);
    return __builtin_bit_cast(v16bf, u);
}

template<bool TO_GLOBAL>
static __device__ __forceinline__ void run_layer(
    const unsigned short* sIn,              // LDS A-tile [M_TILE][STRIDE] bf16
    const float* __restrict__ W,            // [D][D] fp32, row = output neuron
    const float* __restrict__ bias,         // [D] fp32
    unsigned short* sOut,                   // LDS out (bf16, relu) when !TO_GLOBAL
    float* __restrict__ gOut,               // global out (fp32, relu) when TO_GLOBAL
    int rowBase, int nEdges, int wave, int r, int hi)
{
    #pragma unroll
    for (int half = 0; half < 2; ++half) {
        const int nt   = wave + half * 8;      // this wave's output col-tile
        const int ncol = nt * 16 + r;          // lane's output column (C/D: N = lane&15)
        const float bv = bias[ncol];
        const float* wrow = W + (size_t)ncol * D;

        // Hold all B fragments for this n-tile in registers across the M loop.
        v16bf bfr[8];
        #pragma unroll
        for (int ks = 0; ks < 8; ++ks)
            bfr[ks] = load_bfrag_global(wrow, ks * 32 + hi * 8);

        for (int mt = 0; mt < 8; ++mt) {
            v8f c = {};
            #pragma unroll
            for (int ks = 0; ks < 8; ++ks) {
                v16bf a = load_frag_lds(sIn, (mt * 16 + r) * STRIDE + ks * 32 + hi * 8);
                c = __builtin_amdgcn_wmma_f32_16x16x32_bf16(
                        /*neg_a=*/false, a, /*neg_b=*/false, bfr[ks],
                        /*c_mod=*/(short)0, c, /*reuse_a=*/false, /*reuse_b=*/false);
            }
            #pragma unroll
            for (int j = 0; j < 8; ++j) {
                const float v = fmaxf(c[j] + bv, 0.0f);   // single v_max_f32 relu
                const int m = mt * 16 + hi * 8 + j;       // C/D: VGPR j -> M = j + 8*hi
                if (TO_GLOBAL) {
                    const int grow = rowBase + m;
                    if (grow < nEdges) gOut[(size_t)grow * D + ncol] = v;
                } else {
                    sOut[m * STRIDE + ncol] = f2bf(v);
                }
            }
        }
    }
}

__global__ __launch_bounds__(BLOCK) void edgenet_wmma_kernel(
    const float* __restrict__ parent, const float* __restrict__ child,
    const float* __restrict__ W1, const float* __restrict__ b1,
    const float* __restrict__ W2, const float* __restrict__ b2,
    float* __restrict__ out, int nEdges)
{
    __shared__ __align__(16) unsigned short sX[M_TILE * STRIDE];   // 66 KB
    __shared__ __align__(16) unsigned short sH[M_TILE * STRIDE];   // 66 KB
    __shared__ __align__(16) float          sP[D];                 // 1 KB

    const int tid     = threadIdx.x;
    const int rowBase = blockIdx.x * M_TILE;

    if (tid < D) sP[tid] = parent[tid];
    __syncthreads();

    // Stage x = bf16(parent * child) into LDS, coalesced float4 loads.
    for (int it = tid; it < M_TILE * (D / 4); it += BLOCK) {
        const int row  = it >> 6;        // D/4 == 64
        const int c4   = it & 63;
        const int grow = rowBase + row;
        float4 v = make_float4(0.f, 0.f, 0.f, 0.f);
        if (grow < nEdges)
            v = ((const float4*)child)[(size_t)grow * (D / 4) + c4];
        const float4 p = ((const float4*)sP)[c4];
        const unsigned int lo = pack_bf2(v.x * p.x, v.y * p.y);
        const unsigned int hh = pack_bf2(v.z * p.z, v.w * p.w);
        // STRIDE/4 == 66 uint2 per row
        ((uint2*)sX)[(size_t)row * (STRIDE / 4) + c4] = make_uint2(lo, hh);
    }
    __syncthreads();

    const int wave = tid >> 5;
    const int lane = tid & 31;
    const int r    = lane & 15;
    const int hi   = lane >> 4;

    // Layer 1: sX -> sH (bf16, relu)
    run_layer<false>(sX, W1, b1, sH, nullptr, rowBase, nEdges, wave, r, hi);
    __syncthreads();
    // Layer 2: sH -> out (fp32, relu)
    run_layer<true>(sH, W2, b2, nullptr, out, rowBase, nEdges, wave, r, hi);
}

extern "C" void kernel_launch(void* const* d_in, const int* in_sizes, int n_in,
                              void* d_out, int out_size, void* d_ws, size_t ws_size,
                              hipStream_t stream) {
    const float* parent = (const float*)d_in[0];
    const float* child  = (const float*)d_in[1];
    const float* W1     = (const float*)d_in[2];
    const float* b1     = (const float*)d_in[3];
    const float* W2     = (const float*)d_in[4];
    const float* b2     = (const float*)d_in[5];
    float* out = (float*)d_out;

    const int nEdges = in_sizes[1] / D;
    const int grid   = (nEdges + M_TILE - 1) / M_TILE;
    edgenet_wmma_kernel<<<grid, BLOCK, 0, stream>>>(parent, child, W1, b1, W2, b2, out, nEdges);
}
